// LightweightMIL_Conv_65034394796217
// MI455X (gfx1250) — matrix operand order repaired and checked by reading, not compiled, not run on hardware
//
#include <hip/hip_runtime.h>
#include <hip/hip_bf16.h>
#include <math.h>

// ---------- WMMA vector types ----------
typedef __attribute__((ext_vector_type(16))) __bf16         v16bf;
typedef __attribute__((ext_vector_type(8)))  float          v8f;

// Native RNE f32->bf16 (lowers to v_cvt_pk_bf16_f32)
static __device__ __forceinline__ __bf16 f2bf(float f) {
    return (__bf16)f;
}
static __device__ __forceinline__ unsigned short f2bfu(float f) {
    return __builtin_bit_cast(unsigned short, (__bf16)f);
}

// ---------------------------------------------------------------
// K0: pre-swizzle Wd (512x64) and Wa1 (64x32) into bf16 WMMA
//     B-fragment layout: frag f = kstep*NT + ntile,
//     element (k,n): kr=k%32, lane = (n%16) + 16*((kr/8)%2), j = kr%8 + 8*(kr/16)
// ---------------------------------------------------------------
__global__ void k_prep(const float* __restrict__ Wd, const float* __restrict__ Wa1,
                       unsigned short* __restrict__ wdswz, unsigned short* __restrict__ wa1swz) {
    int t = threadIdx.x;
    for (int idx = t; idx < 512 * 64; idx += 256) {
        int k = idx >> 6, n = idx & 63;
        int ks = k >> 5, kr = k & 31;
        int f = ks * 4 + (n >> 4);
        int lane = (n & 15) + (((kr >> 3) & 1) << 4);
        int j = (kr & 7) + ((kr >> 4) << 3);
        wdswz[(f * 32 + lane) * 16 + j] = f2bfu(Wd[idx]);
    }
    for (int idx = t; idx < 64 * 32; idx += 256) {
        int k = idx >> 5, n = idx & 31;
        int ks = k >> 5, kr = k & 31;
        int f = ks * 2 + (n >> 4);
        int lane = (n & 15) + (((kr >> 3) & 1) << 4);
        int j = (kr & 7) + ((kr >> 4) << 3);
        wa1swz[(f * 32 + lane) * 16 + j] = f2bfu(Wa1[idx]);
    }
}

// ---------------------------------------------------------------
// K2: h = x @ Wd + bd  (bf16 WMMA, fp32 accumulate), fused with
//     attention MLP: s = tanh(h@Wa1+ba1)@Wa2+ba2, masked scores.
//     One wave owns a 16-row tile (16 x 64 output). 4 waves/block.
// ---------------------------------------------------------------
__global__ __launch_bounds__(128)
void k_gemm_h_attn(const float* __restrict__ x,  const float* __restrict__ bd,
                   const float* __restrict__ ba1, const float* __restrict__ ba2,
                   const float* __restrict__ Wa2,
                   const unsigned short* __restrict__ wdswz,
                   const unsigned short* __restrict__ wa1swz,
                   float* __restrict__ h_out, float* __restrict__ scores) {
    __shared__ __attribute__((aligned(32))) unsigned short lds_wd[32 * 32 * 16]; // 32KB (half of Wd)
    __shared__ __attribute__((aligned(32))) unsigned short lds_wa1[4 * 32 * 16]; //  4KB
    __shared__ __attribute__((aligned(32))) unsigned short lds_scr[4 * 1024];    //  8KB (per-wave A scratch)

    const int tid  = threadIdx.x;
    const int wave = tid >> 5;
    const int lane = tid & 31;
    const int lm   = lane & 15;
    const int hi   = lane >> 4;

    // Wa1 fragments: fill once, b128 granularity (visibility via phase-0 barrier)
    {
        const uint4* s2 = (const uint4*)wa1swz;
        uint4* d2 = (uint4*)lds_wa1;
        for (int i = tid; i < 256; i += 128) d2[i] = s2[i];
    }

    const int tile = blockIdx.x * 4 + wave;   // 16384 tiles total
    const int r0   = tile * 16;               // global row base (b*4096+n)

    v8f c[4];
#pragma unroll
    for (int t = 0; t < 4; ++t) c[t] = (v8f)0.0f;

    // A-fragment source: lane l -> row r0+(l&15), K-half (l>>4)*8
    const float* xr = x + (size_t)(r0 + lm) * 512 + (hi << 3);

    for (int phase = 0; phase < 2; ++phase) {
        // stage 32 Wd fragments (8 ksteps) into LDS, b128 granularity
        {
            const uint4* src = (const uint4*)wdswz + phase * 2048;
            uint4* dst = (uint4*)lds_wd;
            for (int i = tid; i < 2048; i += 128) dst[i] = src[i];
        }
        __syncthreads();

        for (int ks8 = 0; ks8 < 8; ++ks8) {
            const int ks = phase * 8 + ks8;
            const float* p = xr + ks * 32;
            float4 f0 = *(const float4*)(p);
            float4 f1 = *(const float4*)(p + 4);
            float4 f2 = *(const float4*)(p + 16);
            float4 f3 = *(const float4*)(p + 20);
            if (ks < 15) __builtin_prefetch(p + 32, 0, 3);   // global_prefetch next kstep

            v16bf a;
            a[0]  = f2bf(f0.x); a[1]  = f2bf(f0.y); a[2]  = f2bf(f0.z); a[3]  = f2bf(f0.w);
            a[4]  = f2bf(f1.x); a[5]  = f2bf(f1.y); a[6]  = f2bf(f1.z); a[7]  = f2bf(f1.w);
            a[8]  = f2bf(f2.x); a[9]  = f2bf(f2.y); a[10] = f2bf(f2.z); a[11] = f2bf(f2.w);
            a[12] = f2bf(f3.x); a[13] = f2bf(f3.y); a[14] = f2bf(f3.z); a[15] = f2bf(f3.w);

#pragma unroll
            for (int nt = 0; nt < 4; ++nt) {
                v16bf b = *(const v16bf*)&lds_wd[((ks8 * 4 + nt) * 32 + lane) * 16];
                c[nt] = __builtin_amdgcn_wmma_f32_16x16x32_bf16(
                            false, a, false, b, (short)0, c[nt], false, false);
            }
        }
        __syncthreads();   // all waves done reading before next refill
    }

    // add bd, write h tile, and stash bf16 h in A-fragment layout (per-wave LDS)
    float* hbase = h_out + (size_t)r0 * 64;
    unsigned short* scr = &lds_scr[wave * 1024];
#pragma unroll
    for (int nt = 0; nt < 4; ++nt) {
        float bdv = bd[lm + nt * 16];
#pragma unroll
        for (int v = 0; v < 8; ++v) {
            float val = c[nt][v] + bdv;
            c[nt][v] = val;
            int M = v + 8 * hi;
            int n = lm + nt * 16;
            hbase[(size_t)M * 64 + n] = val;
            // A layout (16-bit, 16x32): kk<8->lane M,j=kk; kk<16->M+16,j=kk-8;
            //                           kk<24->M,j=kk-8;   else M+16,j=kk-16
            int ks2 = n >> 5, kk = n & 31;
            int la, j;
            if (kk < 8)       { la = M;      j = kk;      }
            else if (kk < 16) { la = M + 16; j = kk - 8;  }
            else if (kk < 24) { la = M;      j = kk - 8;  }
            else              { la = M + 16; j = kk - 16; }
            scr[(ks2 * 32 + la) * 16 + j] = f2bfu(val);
        }
    }
    // per-wave DS ops are in-order: stores above are visible to loads below

    // mask partial: each lane covers 4 columns of |h|
    float mp[8], sp[8];
#pragma unroll
    for (int v = 0; v < 8; ++v) {
        mp[v] = fabsf(c[0][v]) + fabsf(c[1][v]) + fabsf(c[2][v]) + fabsf(c[3][v]);
        sp[v] = 0.0f;
    }

    // t = tanh(h @ Wa1 + ba1) : 16x32, two N-tiles, K=64 (2 ksteps)
#pragma unroll
    for (int t2 = 0; t2 < 2; ++t2) {
        v8f tacc = (v8f)0.0f;
#pragma unroll
        for (int ks2 = 0; ks2 < 2; ++ks2) {
            v16bf a2 = *(const v16bf*)&scr[(ks2 * 32 + lane) * 16];
            v16bf b2 = *(const v16bf*)&lds_wa1[((ks2 * 2 + t2) * 32 + lane) * 16];
            tacc = __builtin_amdgcn_wmma_f32_16x16x32_bf16(
                       false, a2, false, b2, (short)0, tacc, false, false);
        }
        float bv = ba1[lm + t2 * 16];
        float wv = Wa2[lm + t2 * 16];
#pragma unroll
        for (int v = 0; v < 8; ++v) sp[v] += tanhf(tacc[v] + bv) * wv;
    }

    // reduce over the 16 lanes of each half (rows 0-7 in lanes 0-15, 8-15 in 16-31)
#pragma unroll
    for (int m = 1; m < 16; m <<= 1) {
#pragma unroll
        for (int v = 0; v < 8; ++v) {
            sp[v] += __shfl_xor(sp[v], m, 16);
            mp[v] += __shfl_xor(mp[v], m, 16);
        }
    }
    if (lm == 0) {
        float b2v = ba2[0];
#pragma unroll
        for (int v = 0; v < 8; ++v) {
            int row = v + 8 * hi;
            float mask = (mp[v] > 0.0f) ? 1.0f : 0.0f;
            scores[r0 + row] = (sp[v] + b2v) * mask - 1e10f * (1.0f - mask);
        }
    }
}

// ---------------------------------------------------------------
// K3: per-batch softmax stats (max, sum of exp)
// ---------------------------------------------------------------
__global__ void k_softmax_stats(const float* __restrict__ scores,
                                float* __restrict__ smax, float* __restrict__ ssum) {
    __shared__ float red[256];
    int b = blockIdx.x, t = threadIdx.x;
    const float* s = scores + (size_t)b * 4096;
    float m = -3.4e38f;
    for (int n = t; n < 4096; n += 256) m = fmaxf(m, s[n]);
    red[t] = m; __syncthreads();
    for (int k = 128; k > 0; k >>= 1) { if (t < k) red[t] = fmaxf(red[t], red[t + k]); __syncthreads(); }
    float bm = red[0]; __syncthreads();
    float sum = 0.0f;
    for (int n = t; n < 4096; n += 256) sum += __expf(s[n] - bm);
    red[t] = sum; __syncthreads();
    for (int k = 128; k > 0; k >>= 1) { if (t < k) red[t] += red[t + k]; __syncthreads(); }
    if (t == 0) { smax[b] = bm; ssum[b] = red[0]; }
}

// ---------------------------------------------------------------
// K4: grouped[b,g,:] = sum_{n in group g} h[b,n,:] * attn[b,n]
// ---------------------------------------------------------------
__global__ void k_segsum(const float* __restrict__ h, const float* __restrict__ scores,
                         const float* __restrict__ smax, const float* __restrict__ ssum,
                         float* __restrict__ grouped) {
    __shared__ float red[256];
    int b = blockIdx.x / 10, g = blockIdx.x % 10;
    int t = threadIdx.x, hd = t & 63, sub = t >> 6;
    int n0 = g * 410, n1 = min(n0 + 410, 4096);
    float bm = smax[b], inv = 1.0f / ssum[b];
    float acc = 0.0f;
    for (int n = n0 + sub; n < n1; n += 4) {
        float w = __expf(scores[(size_t)b * 4096 + n] - bm) * inv;
        acc += h[((size_t)b * 4096 + n) * 64 + hd] * w;
    }
    red[t] = acc; __syncthreads();
    if (t < 64)
        grouped[(size_t)(b * 10 + g) * 64 + t] = red[t] + red[t + 64] + red[t + 128] + red[t + 192];
}

// ---------------------------------------------------------------
// K5: conv1 (10->64ch, K=3, same pad) + 1x1 residual conv
// ---------------------------------------------------------------
__global__ void k_conv1(const float* __restrict__ grouped, const float* __restrict__ c1w,
                        const float* __restrict__ c1b, const float* __restrict__ r1w,
                        const float* __restrict__ r1b,
                        float* __restrict__ z1, float* __restrict__ rz) {
    __shared__ float gs[640];
    int b = blockIdx.x, t = threadIdx.x;
    for (int i = t; i < 640; i += 256) gs[i] = grouped[(size_t)b * 640 + i];
    __syncthreads();
    for (int idx = t; idx < 4096; idx += 256) {
        int o = idx >> 6, l = idx & 63;
        float acc = c1b[o], racc = r1b[o];
        for (int i = 0; i < 10; ++i) {
            const float* w = c1w + (o * 10 + i) * 3;
            float xm = (l > 0)  ? gs[i * 64 + l - 1] : 0.0f;
            float x0 = gs[i * 64 + l];
            float xp = (l < 63) ? gs[i * 64 + l + 1] : 0.0f;
            acc  += xm * w[0] + x0 * w[1] + xp * w[2];
            racc += x0 * r1w[o * 10 + i];
        }
        z1[(size_t)b * 4096 + idx] = acc;
        rz[(size_t)b * 4096 + idx] = racc;
    }
}

// ---------------------------------------------------------------
// K6/K8: per-channel sum & sum-of-squares over (B,L) — deterministic
// ---------------------------------------------------------------
__global__ void k_bnstats(const float* __restrict__ z, float* __restrict__ sum,
                          float* __restrict__ sq) {
    __shared__ float r1[256], r2[256];
    int c = blockIdx.x, t = threadIdx.x;
    float s = 0.0f, q = 0.0f;
    for (int i = t; i < 4096; i += 256) {
        int b = i >> 6, l = i & 63;
        float v = z[((size_t)b * 64 + c) * 64 + l];
        s += v; q += v * v;
    }
    r1[t] = s; r2[t] = q; __syncthreads();
    for (int k = 128; k > 0; k >>= 1) {
        if (t < k) { r1[t] += r1[t + k]; r2[t] += r2[t + k]; }
        __syncthreads();
    }
    if (t == 0) { sum[c] = r1[0]; sq[c] = r2[0]; }
}

// ---------------------------------------------------------------
// K7: y1 = relu(bn1(z1)) + rz ; z2 = conv2(y1) (64->64ch, K=3)
// ---------------------------------------------------------------
__global__ __launch_bounds__(256)
void k_bn1_conv2(const float* __restrict__ z1, const float* __restrict__ rz,
                 const float* __restrict__ bsum, const float* __restrict__ bsq,
                 const float* __restrict__ bn1g, const float* __restrict__ bn1b,
                 const float* __restrict__ c2w, const float* __restrict__ c2b,
                 float* __restrict__ y1, float* __restrict__ z2) {
    __shared__ float ys[4096];
    int b = blockIdx.x, t = threadIdx.x;
    for (int idx = t; idx < 4096; idx += 256) {
        int cch = idx >> 6;
        float mean = bsum[cch] * (1.0f / 4096.0f);
        float var  = bsq[cch] * (1.0f / 4096.0f) - mean * mean;
        float rs   = rsqrtf(var + 1e-5f);
        float v = (z1[(size_t)b * 4096 + idx] - mean) * rs * bn1g[cch] + bn1b[cch];
        v = fmaxf(v, 0.0f) + rz[(size_t)b * 4096 + idx];
        ys[idx] = v;
        y1[(size_t)b * 4096 + idx] = v;
    }
    __syncthreads();
    for (int idx = t; idx < 4096; idx += 256) {
        int o = idx >> 6, l = idx & 63;
        float acc = c2b[o];
        const float* wrow = c2w + o * 192;
        for (int i = 0; i < 64; ++i) {
            float xm = (l > 0)  ? ys[i * 64 + l - 1] : 0.0f;
            float x0 = ys[i * 64 + l];
            float xp = (l < 63) ? ys[i * 64 + l + 1] : 0.0f;
            acc += xm * wrow[i * 3] + x0 * wrow[i * 3 + 1] + xp * wrow[i * 3 + 2];
        }
        z2[(size_t)b * 4096 + idx] = acc;
    }
}

// ---------------------------------------------------------------
// K9: y2 = relu(bn2(z2)) + y1 ; pooled = mean_L ; logits = pooled@Wc+bc
// ---------------------------------------------------------------
__global__ void k_final(const float* __restrict__ z2, const float* __restrict__ y1,
                        const float* __restrict__ b2sum, const float* __restrict__ b2sq,
                        const float* __restrict__ bn2g, const float* __restrict__ bn2b,
                        const float* __restrict__ Wc, const float* __restrict__ bc,
                        float* __restrict__ out) {
    __shared__ float pooled[64];
    int b = blockIdx.x, t = threadIdx.x;  // 64 threads, t = channel
    float mean = b2sum[t] * (1.0f / 4096.0f);
    float var  = b2sq[t] * (1.0f / 4096.0f) - mean * mean;
    float rs   = rsqrtf(var + 1e-5f);
    float g = bn2g[t], beta = bn2b[t];
    const float* zr = z2 + ((size_t)b * 64 + t) * 64;
    const float* yr = y1 + ((size_t)b * 64 + t) * 64;
    float acc = 0.0f;
    for (int l = 0; l < 64; ++l)
        acc += fmaxf((zr[l] - mean) * rs * g + beta, 0.0f) + yr[l];
    pooled[t] = acc * (1.0f / 64.0f);
    __syncthreads();
    if (t < 2) {
        float lg = bc[t];
        for (int c = 0; c < 64; ++c) lg += pooled[c] * Wc[c * 2 + t];
        out[b * 2 + t] = lg;
    }
}

// ---------------------------------------------------------------
extern "C" void kernel_launch(void* const* d_in, const int* in_sizes, int n_in,
                              void* d_out, int out_size, void* d_ws, size_t ws_size,
                              hipStream_t stream) {
    const float* x    = (const float*)d_in[0];
    const float* Wd   = (const float*)d_in[1];
    const float* bd   = (const float*)d_in[2];
    const float* Wa1  = (const float*)d_in[3];
    const float* ba1  = (const float*)d_in[4];
    const float* Wa2  = (const float*)d_in[5];
    const float* ba2  = (const float*)d_in[6];
    const float* c1w  = (const float*)d_in[7];
    const float* c1b  = (const float*)d_in[8];
    const float* bn1g = (const float*)d_in[9];
    const float* bn1b = (const float*)d_in[10];
    const float* r1w  = (const float*)d_in[11];
    const float* r1b  = (const float*)d_in[12];
    const float* c2w  = (const float*)d_in[13];
    const float* c2b  = (const float*)d_in[14];
    const float* bn2g = (const float*)d_in[15];
    const float* bn2b = (const float*)d_in[16];
    const float* Wc   = (const float*)d_in[17];
    const float* bc   = (const float*)d_in[18];
    (void)in_sizes; (void)n_in; (void)out_size; (void)ws_size;

    char* ws = (char*)d_ws;
    const size_t OFF_WD   = 0;                              // 65536 B  (Wd bf16 swizzled)
    const size_t OFF_WA1  = OFF_WD  + 65536;                // 4096 B
    const size_t OFF_H    = OFF_WA1 + 4096;                 // 67108864 B (h fp32)
    const size_t OFF_SC   = OFF_H   + 67108864;             // 1048576 B  (scores)
    const size_t OFF_SMAX = OFF_SC  + 1048576;              // 256 B
    const size_t OFF_SSUM = OFF_SMAX + 256;                 // 256 B
    const size_t OFF_GRP  = OFF_SSUM + 256;                 // 163840 B
    const size_t OFF_Z1   = OFF_GRP + 163840;               // 1048576 B
    const size_t OFF_RZ   = OFF_Z1  + 1048576;
    const size_t OFF_Y1   = OFF_RZ  + 1048576;
    const size_t OFF_Z2   = OFF_Y1  + 1048576;
    const size_t OFF_B1S  = OFF_Z2  + 1048576;
    const size_t OFF_B1Q  = OFF_B1S + 256;
    const size_t OFF_B2S  = OFF_B1Q + 256;
    const size_t OFF_B2Q  = OFF_B2S + 256;

    unsigned short* wdswz  = (unsigned short*)(ws + OFF_WD);
    unsigned short* wa1swz = (unsigned short*)(ws + OFF_WA1);
    float* h      = (float*)(ws + OFF_H);
    float* scores = (float*)(ws + OFF_SC);
    float* smax   = (float*)(ws + OFF_SMAX);
    float* ssum   = (float*)(ws + OFF_SSUM);
    float* grp    = (float*)(ws + OFF_GRP);
    float* z1     = (float*)(ws + OFF_Z1);
    float* rz     = (float*)(ws + OFF_RZ);
    float* y1     = (float*)(ws + OFF_Y1);
    float* z2     = (float*)(ws + OFF_Z2);
    float* b1s    = (float*)(ws + OFF_B1S);
    float* b1q    = (float*)(ws + OFF_B1Q);
    float* b2s    = (float*)(ws + OFF_B2S);
    float* b2q    = (float*)(ws + OFF_B2Q);

    k_prep<<<1, 256, 0, stream>>>(Wd, Wa1, wdswz, wa1swz);
    k_gemm_h_attn<<<4096, 128, 0, stream>>>(x, bd, ba1, ba2, Wa2, wdswz, wa1swz, h, scores);
    k_softmax_stats<<<64, 256, 0, stream>>>(scores, smax, ssum);
    k_segsum<<<640, 256, 0, stream>>>(h, scores, smax, ssum, grp);
    k_conv1<<<64, 256, 0, stream>>>(grp, c1w, c1b, r1w, r1b, z1, rz);
    k_bnstats<<<64, 256, 0, stream>>>(z1, b1s, b1q);
    k_bn1_conv2<<<64, 256, 0, stream>>>(z1, rz, b1s, b1q, bn1g, bn1b, c2w, c2b, y1, z2);
    k_bnstats<<<64, 256, 0, stream>>>(z2, b2s, b2q);
    k_final<<<64, 64, 0, stream>>>(z2, y1, b2s, b2q, bn2g, bn2b, Wc, bc, (float*)d_out);
}
